// BiLSTM_57913339019771
// MI455X (gfx1250) — matrix operand order repaired and checked by reading, not compiled, not run on hardware
//
#include <hip/hip_runtime.h>
#include <hip/hip_bf16.h>

// ---------------------------------------------------------------------------
// BiLSTM (2 layers, B=32, T=512, D=256, H=512) for gfx1250 (MI455X, wave32).
// bf16 WMMA (v_wmma_f32_16x16x32_bf16) for all GEMMs, f32 accumulation.
// - Input projections: register-blocked WMMA GEMM (2 M-tiles x 4 N-tiles/wave).
// - Recurrence: persistent kernel, Wh held entirely in VGPRs (16 B-fragments
//   per wave), f32 cell state in registers, per-direction grid barrier/step.
// ---------------------------------------------------------------------------

typedef __attribute__((ext_vector_type(16))) __bf16 v16bf;
typedef __attribute__((ext_vector_type(8)))  float  v8f;

union Frag16 { v16bf v; unsigned u[8]; };

__device__ inline unsigned short f2bf(float x) {
  union { float f; unsigned u; } cv; cv.f = x;
  unsigned r = cv.u + 0x7FFFu + ((cv.u >> 16) & 1u);   // round-to-nearest-even
  return (unsigned short)(r >> 16);
}

__device__ inline float sig_(float x) { return 1.0f / (1.0f + __expf(-x)); }
__device__ inline float tanh_(float x) {
  float t = __expf(-2.0f * fabsf(x));
  float r = (1.0f - t) / (1.0f + t);
  return (x < 0.0f) ? -r : r;
}

// A-matrix (bf16 16x32, MxK) per-lane K offset for packed u32 word q:
// lanes 0-15 hold K {0..15}; lanes 16-31 hold K {16..31} (matches packed B).
__device__ inline int a_koff(int lane, int q) {
  int base = 2 * q;
  return base + ((lane & 16) ? 16 : 0);
}

// ---------------------------------------------------------------------------
// f32 -> bf16 elementwise convert
// ---------------------------------------------------------------------------
__global__ __launch_bounds__(256) void f32_to_bf16_kernel(
    const float* __restrict__ in, unsigned short* __restrict__ out, int n) {
  int i = blockIdx.x * blockDim.x + threadIdx.x;
  if (i < n) out[i] = f2bf(in[i]);
}

// ---------------------------------------------------------------------------
// Pack a [K, N=2048] f32 weight matrix into bf16 B-fragment layout:
// fragment (kb, nt): 32 lanes x 16 bf16. Lane l -> column nt*16 + (l&15),
// K-half (l&16), word q holds K = kb*32 + khalf*16 + 2q(+1).
// ---------------------------------------------------------------------------
__global__ __launch_bounds__(256) void pack_b_kernel(
    const float* __restrict__ W, unsigned short* __restrict__ out, int KB, int N) {
  int tid = blockIdx.x * blockDim.x + threadIdx.x;
  int NT = N >> 4;
  int total = KB * NT * 32;
  if (tid >= total) return;
  int lane = tid & 31;
  int nt = (tid >> 5) % NT;
  int kb = tid / (32 * NT);
  int n = nt * 16 + (lane & 15);
  int kbase = kb * 32 + ((lane & 16) ? 16 : 0);
#pragma unroll
  for (int e = 0; e < 16; ++e)
    out[(size_t)tid * 16 + e] = f2bf(W[(size_t)(kbase + e) * N + n]);
}

// ---------------------------------------------------------------------------
// Input projection: xw = A @ Wi + bias, A: bf16 [B][T][Din] row-major.
// Register-blocked: each wave computes one t (both 16-row batch halves) x
// 4 consecutive N-tiles => 8 accumulators; per K-step 2 A-frags + 4 B-frags
// feed 8 WMMAs. Output in C-fragment-major layout:
// xw[t][nt(128)][mtile(2)][lane(32)][r(8)]  (32B contiguous per lane).
// ---------------------------------------------------------------------------
__global__ __launch_bounds__(256) void gemm_xw_kernel(
    const unsigned short* __restrict__ Abf, const unsigned short* __restrict__ wP,
    const float* __restrict__ bias, float* __restrict__ xw_out, int T, int Din) {
  const int NT = 128;                       // 2048 / 16
  const int NG = NT / 4;                    // 32 groups of 4 N-tiles
  int gid = blockIdx.x * blockDim.x + threadIdx.x;
  int wid = gid >> 5;
  int lane = gid & 31;
  if (wid >= T * NG) return;
  int ng = wid % NG;
  int t = wid / NG;
  int nt0 = ng * 4;
  int KB = Din >> 5;

  v8f acc[2][4];
#pragma unroll
  for (int mt = 0; mt < 2; ++mt)
#pragma unroll
    for (int n = 0; n < 4; ++n) {
      float bv = bias[(nt0 + n) * 16 + (lane & 15)];
      v8f a0 = {bv, bv, bv, bv, bv, bv, bv, bv};
      acc[mt][n] = a0;
    }

  const unsigned* arow0 =
      (const unsigned*)(Abf + ((size_t)(0 + (lane & 15)) * T + t) * Din);
  const unsigned* arow1 =
      (const unsigned*)(Abf + ((size_t)(16 + (lane & 15)) * T + t) * Din);

  for (int kb = 0; kb < KB; ++kb) {
    Frag16 a[2];
#pragma unroll
    for (int q = 0; q < 8; ++q) {
      int idx = (kb * 32 + a_koff(lane, q)) >> 1;
      a[0].u[q] = arow0[idx];
      a[1].u[q] = arow1[idx];
    }
    Frag16 b[4];
#pragma unroll
    for (int n = 0; n < 4; ++n) {
      const unsigned* bp =
          (const unsigned*)wP + ((size_t)(kb * NT + nt0 + n) * 32 + lane) * 8;
#pragma unroll
      for (int q = 0; q < 8; ++q) b[n].u[q] = bp[q];
    }
#pragma unroll
    for (int mt = 0; mt < 2; ++mt)
#pragma unroll
      for (int n = 0; n < 4; ++n)
        acc[mt][n] = __builtin_amdgcn_wmma_f32_16x16x32_bf16(
            false, a[mt].v, false, b[n].v, (short)0, acc[mt][n], false, false);
  }

#pragma unroll
  for (int mt = 0; mt < 2; ++mt)
#pragma unroll
    for (int n = 0; n < 4; ++n) {
      float* dst =
          xw_out + ((((size_t)t * NT + nt0 + n) * 2 + mt) * 32 + lane) * 8;
#pragma unroll
      for (int r = 0; r < 8; ++r) dst[r] = acc[mt][n][r];
    }
}

// ---------------------------------------------------------------------------
// Zero the double-buffered h state + per-direction barrier counters.
// ---------------------------------------------------------------------------
__global__ __launch_bounds__(256) void init_state_kernel(
    unsigned short* __restrict__ hstate, unsigned* __restrict__ cnt, int n) {
  int i = blockIdx.x * blockDim.x + threadIdx.x;
  if (i < n) hstate[i] = 0;
  if (i < 64) cnt[i] = 0;
}

// ---------------------------------------------------------------------------
// Persistent recurrent kernel. Grid = 128 WGs (2 dirs x 2 batch-halves x 32
// h-column tiles), 128 threads (4 waves). Waves split K=512 into 4 chunks;
// each wave's 16 Wh B-fragments (4 kb x 4 gates) are loaded ONCE into VGPRs
// and stay register-resident for all T steps. Partials reduced through LDS;
// wave 0 applies gates, keeps c in registers, writes h (double-buffered).
// Per-direction device barrier (monotonic atomic counter, 64 WGs each).
// ---------------------------------------------------------------------------
__global__ __launch_bounds__(128) void lstm_recur_kernel(
    const float* __restrict__ xw,            // [2][T][128][2][32][8] f32 frag-major
    const unsigned short* __restrict__ whP,  // [2][16*128*512] packed bf16
    unsigned short* __restrict__ hstate,     // [2 phases][2 dirs][32][512] bf16
    unsigned short* __restrict__ hid_bf,     // [B][T][1024] bf16 (layer0) or null
    float* __restrict__ out_f32,             // [B][T][1024] f32 (layer1) or null
    int T, unsigned* __restrict__ cnt) {
  __shared__ float red[4][4][32][8];
  const int j = blockIdx.x & 31;
  const int mtile = (blockIdx.x >> 5) & 1;
  const int d = blockIdx.x >> 6;
  const int w = threadIdx.x >> 5;
  const int lane = threadIdx.x & 31;
  const float* xw_d = xw + (size_t)d * T * 32 * 2048;
  const unsigned short* wh_d = whP + (size_t)d * 16 * 128 * 512;
  unsigned* mycnt = cnt + d * 16;           // 64B apart per direction
  const int m = mtile * 16 + (lane & 15);   // A-row (batch index) for this lane
  const int rbase = (lane & 16) ? 8 : 0;    // C-layout M offset

  // Load this wave's 16 Wh B-fragments into registers (persist across steps).
  Frag16 bw[16];
#pragma unroll
  for (int kk = 0; kk < 4; ++kk)
#pragma unroll
    for (int g = 0; g < 4; ++g) {
      const int kb = w * 4 + kk;
      const unsigned* bp =
          (const unsigned*)wh_d + ((size_t)(kb * 128 + g * 32 + j) * 32 + lane) * 8;
#pragma unroll
      for (int q = 0; q < 8; ++q) bw[kk * 4 + g].u[q] = bp[q];
    }

  v8f c = {0.f, 0.f, 0.f, 0.f, 0.f, 0.f, 0.f, 0.f};

  for (int step = 0; step < T; ++step) {
    const int tt = (d == 0) ? step : (T - 1 - step);
    const unsigned short* h_rd = hstate + (size_t)(step & 1) * (2 * 32 * 512);
    unsigned short* h_wr = hstate + (size_t)((step & 1) ^ 1) * (2 * 32 * 512);

    v8f acc[4];
    if (w == 0) {
      // C initializer = precomputed input projection xw[tt] (fragment-major)
#pragma unroll
      for (int g = 0; g < 4; ++g) {
        const float* src =
            xw_d + ((((size_t)tt * 128 + (g * 32 + j)) * 2 + mtile) * 32 + lane) * 8;
#pragma unroll
        for (int r = 0; r < 8; ++r) acc[g][r] = src[r];
      }
      // Prefetch next step's xw tiles behind the compute+barrier phase.
      const int tn = (d == 0) ? (step + 1) : (T - 2 - step);
      if (tn >= 0 && tn < T) {
#pragma unroll
        for (int g = 0; g < 4; ++g)
          __builtin_prefetch(
              xw_d + ((((size_t)tn * 128 + (g * 32 + j)) * 2 + mtile) * 32 + lane) * 8,
              0, 1);
      }
    } else {
      v8f z = {0.f, 0.f, 0.f, 0.f, 0.f, 0.f, 0.f, 0.f};
#pragma unroll
      for (int g = 0; g < 4; ++g) acc[g] = z;
    }

    // Partial h @ Wh over this wave's K chunk (4 k-blocks of 32)
#pragma unroll
    for (int kk = 0; kk < 4; ++kk) {
      const int kb = w * 4 + kk;
      Frag16 a;
      const unsigned* arow = (const unsigned*)(h_rd + (size_t)(d * 32 + m) * 512);
#pragma unroll
      for (int q = 0; q < 8; ++q) a.u[q] = arow[(kb * 32 + a_koff(lane, q)) >> 1];
#pragma unroll
      for (int g = 0; g < 4; ++g)
        acc[g] = __builtin_amdgcn_wmma_f32_16x16x32_bf16(
            false, a.v, false, bw[kk * 4 + g].v, (short)0, acc[g], false, false);
    }

    if (w != 0) {
#pragma unroll
      for (int g = 0; g < 4; ++g)
#pragma unroll
        for (int r = 0; r < 8; ++r) red[w][g][lane][r] = acc[g][r];
    }
    __syncthreads();

    if (w == 0) {
#pragma unroll
      for (int g = 0; g < 4; ++g)
#pragma unroll
        for (int ww = 1; ww < 4; ++ww)
#pragma unroll
          for (int r = 0; r < 8; ++r) acc[g][r] += red[ww][g][lane][r];

#pragma unroll
      for (int r = 0; r < 8; ++r) {
        float ig = sig_(acc[0][r]);
        float fg = sig_(acc[1][r]);
        float gg = tanh_(acc[2][r]);
        float og = sig_(acc[3][r]);
        float cn = fg * c[r] + ig * gg;
        c[r] = cn;
        float hh = og * tanh_(cn);
        const int b = mtile * 16 + rbase + r;
        const int col = j * 16 + (lane & 15);
        h_wr[(size_t)(d * 32 + b) * 512 + col] = f2bf(hh);
        if (hid_bf) hid_bf[((size_t)b * T + tt) * 1024 + d * 512 + col] = f2bf(hh);
        if (out_f32) out_f32[((size_t)b * T + tt) * 1024 + d * 512 + col] = hh;
      }
    }

    __threadfence();
    __syncthreads();
    if (threadIdx.x == 0) {
      atomicAdd(mycnt, 1u);
      const unsigned tgt = 64u * (unsigned)(step + 1);   // 64 WGs per direction
      while (__hip_atomic_load(mycnt, __ATOMIC_ACQUIRE, __HIP_MEMORY_SCOPE_AGENT) < tgt)
        __builtin_amdgcn_s_sleep(1);
    }
    __syncthreads();
  }
}

// ---------------------------------------------------------------------------
// Host-side orchestration
// ---------------------------------------------------------------------------
extern "C" void kernel_launch(void* const* d_in, const int* in_sizes, int n_in,
                              void* d_out, int out_size, void* d_ws, size_t ws_size,
                              hipStream_t stream) {
  (void)in_sizes; (void)n_in; (void)out_size; (void)ws_size;
  const int B = 32, T = 512, D = 256;

  const float* x     = (const float*)d_in[0];
  const float* Wi_f0 = (const float*)d_in[1];
  const float* Wh_f0 = (const float*)d_in[2];
  const float* b_f0  = (const float*)d_in[3];
  const float* Wi_b0 = (const float*)d_in[4];
  const float* Wh_b0 = (const float*)d_in[5];
  const float* b_b0  = (const float*)d_in[6];
  const float* Wi_f1 = (const float*)d_in[7];
  const float* Wh_f1 = (const float*)d_in[8];
  const float* b_f1  = (const float*)d_in[9];
  const float* Wi_b1 = (const float*)d_in[10];
  const float* Wh_b1 = (const float*)d_in[11];
  const float* b_b1  = (const float*)d_in[12];

  char* ws = (char*)d_ws;
  size_t off = 0;
  auto take = [&](size_t bytes) -> char* {
    char* p = ws + off;
    off += (bytes + 255) & ~(size_t)255;
    return p;
  };
  unsigned*       cnt    = (unsigned*)take(256);
  unsigned short* hstate = (unsigned short*)take((size_t)2 * 2 * 32 * 512 * 2);
  unsigned short* x_bf   = (unsigned short*)take((size_t)B * T * D * 2);
  unsigned short* hid_bf = (unsigned short*)take((size_t)B * T * 1024 * 2);
  float*          xw     = (float*)take((size_t)2 * T * 32 * 2048 * 4);
  unsigned short* wi0    = (unsigned short*)take((size_t)2 * 8 * 128 * 512 * 2);
  unsigned short* wh0    = (unsigned short*)take((size_t)2 * 16 * 128 * 512 * 2);
  unsigned short* wi1    = (unsigned short*)take((size_t)2 * 32 * 128 * 512 * 2);
  unsigned short* wh1    = (unsigned short*)take((size_t)2 * 16 * 128 * 512 * 2);

  // x -> bf16
  {
    int n = B * T * D;
    f32_to_bf16_kernel<<<(n + 255) / 256, 256, 0, stream>>>(x, x_bf, n);
  }
  // Pack weights into WMMA B-fragment layout (bf16)
  pack_b_kernel<<<(8 * 128 * 32 + 255) / 256, 256, 0, stream>>>(Wi_f0, wi0, 8, 2048);
  pack_b_kernel<<<(8 * 128 * 32 + 255) / 256, 256, 0, stream>>>(
      Wi_b0, wi0 + (size_t)8 * 128 * 512, 8, 2048);
  pack_b_kernel<<<(16 * 128 * 32 + 255) / 256, 256, 0, stream>>>(Wh_f0, wh0, 16, 2048);
  pack_b_kernel<<<(16 * 128 * 32 + 255) / 256, 256, 0, stream>>>(
      Wh_b0, wh0 + (size_t)16 * 128 * 512, 16, 2048);
  pack_b_kernel<<<(32 * 128 * 32 + 255) / 256, 256, 0, stream>>>(Wi_f1, wi1, 32, 2048);
  pack_b_kernel<<<(32 * 128 * 32 + 255) / 256, 256, 0, stream>>>(
      Wi_b1, wi1 + (size_t)32 * 128 * 512, 32, 2048);
  pack_b_kernel<<<(16 * 128 * 32 + 255) / 256, 256, 0, stream>>>(Wh_f1, wh1, 16, 2048);
  pack_b_kernel<<<(16 * 128 * 32 + 255) / 256, 256, 0, stream>>>(
      Wh_b1, wh1 + (size_t)16 * 128 * 512, 16, 2048);

  // GEMM launch: T*32 waves (one per t x 4-Ntile group), 8 waves per block.
  const int gemmBlocks = (T * 32 * 32) / 256;
  const size_t xwDirStride = (size_t)T * 32 * 2048;

  // ----- Layer 0 -----
  gemm_xw_kernel<<<gemmBlocks, 256, 0, stream>>>(x_bf, wi0, b_f0, xw, T, 256);
  gemm_xw_kernel<<<gemmBlocks, 256, 0, stream>>>(
      x_bf, wi0 + (size_t)8 * 128 * 512, b_b0, xw + xwDirStride, T, 256);
  init_state_kernel<<<256, 256, 0, stream>>>(hstate, cnt, 2 * 2 * 32 * 512);
  lstm_recur_kernel<<<128, 128, 0, stream>>>(xw, wh0, hstate, hid_bf, nullptr, T, cnt);

  // ----- Layer 1 -----
  gemm_xw_kernel<<<gemmBlocks, 256, 0, stream>>>(hid_bf, wi1, b_f1, xw, T, 1024);
  gemm_xw_kernel<<<gemmBlocks, 256, 0, stream>>>(
      hid_bf, wi1 + (size_t)32 * 128 * 512, b_b1, xw + xwDirStride, T, 1024);
  init_state_kernel<<<256, 256, 0, stream>>>(hstate, cnt, 2 * 2 * 32 * 512);
  lstm_recur_kernel<<<128, 128, 0, stream>>>(xw, wh1, hstate, nullptr, (float*)d_out, T, cnt);
}
